// MultiHeadSelfAttention_71829033059057
// MI455X (gfx1250) — matrix operand order repaired and checked by reading, not compiled, not run on hardware
//
#include <hip/hip_runtime.h>
#include <hip/hip_bf16.h>

#define B_   2
#define T_   2048
#define C_   1024
#define H_   16
#define DH_  64
#define M_   (B_ * T_)   // 4096 rows of x

typedef __attribute__((ext_vector_type(16))) _Float16 v16h;
typedef __attribute__((ext_vector_type(8)))  _Float16 v8h;
typedef __attribute__((ext_vector_type(8)))  float    v8f;

__device__ __forceinline__ v8f wmma_f16(v16h a, v16h b, v8f c) {
  // D = A(16x32 f16) * B(32x16 f16) + C(16x16 f32)
  return __builtin_amdgcn_wmma_f32_16x16x32_f16(
      false, a, false, b, (short)0, c, false, false);
}

// A-fragment (16x32, MxK) from row-major f16 matrix; base = tile origin, ld = row stride.
// Lane L: row = L&15; elems 0..7 -> K = (L>>4)*8 + i ; elems 8..15 -> K = 16 + (L>>4)*8 + (i-8)
__device__ __forceinline__ v16h load_frag_a(const _Float16* base, int ld, int lane) {
  const _Float16* p = base + (size_t)(lane & 15) * ld + ((lane >> 4) << 3);
  v8h lo = *(const v8h*)(p);
  v8h hi = *(const v8h*)(p + 16);
  v16h r;
#pragma unroll
  for (int i = 0; i < 8; ++i) { r[i] = lo[i]; r[i + 8] = hi[i]; }
  return r;
}

// B-fragment (32x16, KxN) loaded from TRANSPOSED storage bt[n][k], row stride ld.
// Lane L: col N = L&15; elems 0..7 -> K = (L>>4)*8 + i ; elems 8..15 -> K = 16 + (L>>4)*8 + (i-8)
__device__ __forceinline__ v16h load_frag_bt(const _Float16* bt, int ld, int lane) {
  const _Float16* p = bt + (size_t)(lane & 15) * ld + ((lane >> 4) << 3);
  v8h lo = *(const v8h*)(p);
  v8h hi = *(const v8h*)(p + 16);
  v16h r;
#pragma unroll
  for (int i = 0; i < 8; ++i) { r[i] = lo[i]; r[i + 8] = hi[i]; }
  return r;
}

// ---------------- conversion kernels ----------------
__global__ void cvt_f32_to_f16_kernel(const float* __restrict__ in,
                                      _Float16* __restrict__ out, size_t n) {
  for (size_t i = (size_t)blockIdx.x * blockDim.x + threadIdx.x; i < n;
       i += (size_t)gridDim.x * blockDim.x)
    out[i] = (_Float16)in[i];
}

// in: K x N (row-major f32)  ->  out: N x K (row-major f16)  [out[n][k] = in[k][n]]
__global__ void transpose_cvt_kernel(const float* __restrict__ in,
                                     _Float16* __restrict__ out, int K, int N) {
  size_t total = (size_t)K * N;
  for (size_t i = (size_t)blockIdx.x * blockDim.x + threadIdx.x; i < total;
       i += (size_t)gridDim.x * blockDim.x) {
    size_t k = i / N, n = i % N;
    out[n * (size_t)K + k] = (_Float16)in[i];
  }
}

// ---------------- QKV projection GEMM ----------------
// C[m,n] = xh[m,:] . wT[n,:] + bias[n];  M=4096, N=3072, K=1024
// Block: 256 thr (8 waves), block tile 64(M) x 256(N); wave tile 32 x 64.
__global__ __launch_bounds__(256)
void qkv_gemm_kernel(const _Float16* __restrict__ xh,
                     const _Float16* __restrict__ wT,
                     const float* __restrict__ bias,
                     _Float16* __restrict__ qh,
                     _Float16* __restrict__ kh,
                     _Float16* __restrict__ vTh) {
  const int lane = threadIdx.x & 31;
  const int wave = threadIdx.x >> 5;
  const int m_w = blockIdx.y * 64 + (wave & 1) * 32;
  const int n_w = blockIdx.x * 256 + (wave >> 1) * 64;

  v8f acc[2][4];
#pragma unroll
  for (int mi = 0; mi < 2; ++mi)
#pragma unroll
    for (int ni = 0; ni < 4; ++ni)
#pragma unroll
      for (int r = 0; r < 8; ++r) acc[mi][ni][r] = 0.0f;

  for (int k0 = 0; k0 < C_; k0 += 32) {
    v16h a0 = load_frag_a(xh + (size_t)m_w * C_ + k0, C_, lane);
    v16h a1 = load_frag_a(xh + (size_t)(m_w + 16) * C_ + k0, C_, lane);
#pragma unroll
    for (int ni = 0; ni < 4; ++ni) {
      v16h bf = load_frag_bt(wT + (size_t)(n_w + ni * 16) * C_ + k0, C_, lane);
      acc[0][ni] = wmma_f16(a0, bf, acc[0][ni]);
      acc[1][ni] = wmma_f16(a1, bf, acc[1][ni]);
    }
  }

  const int half = (lane >> 4) << 3;   // rows r (lanes 0-15) / 8+r (lanes 16-31)
  const int col_l = lane & 15;
#pragma unroll
  for (int mi = 0; mi < 2; ++mi)
#pragma unroll
    for (int ni = 0; ni < 4; ++ni)
#pragma unroll
      for (int r = 0; r < 8; ++r) {
        int m = m_w + mi * 16 + r + half;
        int n = n_w + ni * 16 + col_l;
        float v = acc[mi][ni][r] + bias[n];
        _Float16 hv = (_Float16)v;
        int g = n >> 10;            // 0=q 1=k 2=v
        int cc = n & (C_ - 1);
        int hh = cc >> 6, d = cc & 63;
        int bb = m >> 11, t = m & (T_ - 1);
        size_t bh = (size_t)(bb * H_ + hh);
        if (g == 0)      qh[(bh * T_ + t) * DH_ + d] = hv;          // (B,H,T,Dh)
        else if (g == 1) kh[(bh * T_ + t) * DH_ + d] = hv;          // (B,H,T,Dh)
        else             vTh[(bh * DH_ + d) * T_ + t] = hv;         // (B,H,Dh,T)
      }
}

// ---------------- causal flash attention ----------------
// One wave handles 16 query rows; block = 8 waves = 128 rows.
// grid.x = B*H, grid.y = T/128
__global__ __launch_bounds__(256)
void attn_kernel(const _Float16* __restrict__ qh,
                 const _Float16* __restrict__ kh,
                 const _Float16* __restrict__ vTh,
                 _Float16* __restrict__ oh) {
  __shared__ __align__(32) _Float16 p_lds[8][16 * 32];  // per-wave P tile (16x32)

  const int lane = threadIdx.x & 31;
  const int wave = threadIdx.x >> 5;
  const int bh = blockIdx.x;
  const int q0 = blockIdx.y * 128 + wave * 16;

  const _Float16* Qb = qh + ((size_t)bh * T_ + q0) * DH_;
  const _Float16* Kb = kh + (size_t)bh * T_ * DH_;
  const _Float16* Vb = vTh + (size_t)bh * DH_ * T_;
  _Float16* pw = &p_lds[wave][0];

  const v16h qa0 = load_frag_a(Qb, DH_, lane);
  const v16h qa1 = load_frag_a(Qb + 32, DH_, lane);

  v8f O[4];
  float mrow[8], lrow[8];
#pragma unroll
  for (int nf = 0; nf < 4; ++nf)
#pragma unroll
    for (int r = 0; r < 8; ++r) O[nf][r] = 0.0f;
#pragma unroll
  for (int r = 0; r < 8; ++r) { mrow[r] = -1e30f; lrow[r] = 0.0f; }

  const int half = (lane >> 4) << 3;
  const int col_l = lane & 15;
  const int kend = ((q0 + 16) + 31) & ~31;   // causal: keys < q0+16, rounded to step

  for (int kb = 0; kb < kend; kb += 32) {
    v8f sf[2];
#pragma unroll
    for (int c = 0; c < 2; ++c) {
      const _Float16* Kt = Kb + (size_t)(kb + c * 16) * DH_;
      v8f s;
#pragma unroll
      for (int r = 0; r < 8; ++r) s[r] = 0.0f;
      s = wmma_f16(qa0, load_frag_bt(Kt, DH_, lane), s);
      s = wmma_f16(qa1, load_frag_bt(Kt + 32, DH_, lane), s);
#pragma unroll
      for (int r = 0; r < 8; ++r) s[r] *= 0.125f;  // 1/sqrt(64)
      if (kb + 31 > q0) {                           // diagonal block: causal mask
        int key = kb + c * 16 + col_l;
#pragma unroll
        for (int r = 0; r < 8; ++r)
          if (key > q0 + r + half) s[r] = -1e30f;
      }
      sf[c] = s;
    }

    float alpha[8];
#pragma unroll
    for (int r = 0; r < 8; ++r) {
      float v = fmaxf(sf[0][r], sf[1][r]);
      v = fmaxf(v, __shfl_xor(v, 1, 32));
      v = fmaxf(v, __shfl_xor(v, 2, 32));
      v = fmaxf(v, __shfl_xor(v, 4, 32));
      v = fmaxf(v, __shfl_xor(v, 8, 32));   // masks<16 never cross the 16-lane halves
      float mn = fmaxf(mrow[r], v);
      alpha[r] = __expf(mrow[r] - mn);
      mrow[r] = mn;
    }

#pragma unroll
    for (int r = 0; r < 8; ++r) {
      float p0 = __expf(sf[0][r] - mrow[r]);
      float p1 = __expf(sf[1][r] - mrow[r]);
      pw[(r + half) * 32 + col_l] = (_Float16)p0;        // C-layout -> LDS tile
      pw[(r + half) * 32 + 16 + col_l] = (_Float16)p1;
      float ps = p0 + p1;
      ps += __shfl_xor(ps, 1, 32);
      ps += __shfl_xor(ps, 2, 32);
      ps += __shfl_xor(ps, 4, 32);
      ps += __shfl_xor(ps, 8, 32);
      lrow[r] = lrow[r] * alpha[r] + ps;
    }

#pragma unroll
    for (int nf = 0; nf < 4; ++nf)
#pragma unroll
      for (int r = 0; r < 8; ++r) O[nf][r] *= alpha[r];

    v16h pa = load_frag_a(pw, 32, lane);   // same-wave LDS: HW keeps DS ops in order
#pragma unroll
    for (int nf = 0; nf < 4; ++nf) {
      const _Float16* Vt = Vb + (size_t)(nf * 16) * T_ + kb;
      O[nf] = wmma_f16(pa, load_frag_bt(Vt, T_, lane), O[nf]);
    }
  }

  const int bb = bh >> 4, hh = bh & 15;
#pragma unroll
  for (int nf = 0; nf < 4; ++nf)
#pragma unroll
    for (int r = 0; r < 8; ++r) {
      float v = O[nf][r] / lrow[r];
      int row = q0 + r + half;
      oh[((size_t)bb * T_ + row) * C_ + hh * DH_ + nf * 16 + col_l] = (_Float16)v;
    }
}

// ---------------- output projection GEMM ----------------
// out[m,n] = ah[m,:] . wT[n,:] + bias[n];  M=4096, N=1024, K=1024, f32 out
__global__ __launch_bounds__(256)
void out_gemm_kernel(const _Float16* __restrict__ ah,
                     const _Float16* __restrict__ wT,
                     const float* __restrict__ bias,
                     float* __restrict__ out) {
  const int lane = threadIdx.x & 31;
  const int wave = threadIdx.x >> 5;
  const int m_w = blockIdx.y * 64 + (wave & 1) * 32;
  const int n_w = blockIdx.x * 256 + (wave >> 1) * 64;

  v8f acc[2][4];
#pragma unroll
  for (int mi = 0; mi < 2; ++mi)
#pragma unroll
    for (int ni = 0; ni < 4; ++ni)
#pragma unroll
      for (int r = 0; r < 8; ++r) acc[mi][ni][r] = 0.0f;

  for (int k0 = 0; k0 < C_; k0 += 32) {
    v16h a0 = load_frag_a(ah + (size_t)m_w * C_ + k0, C_, lane);
    v16h a1 = load_frag_a(ah + (size_t)(m_w + 16) * C_ + k0, C_, lane);
#pragma unroll
    for (int ni = 0; ni < 4; ++ni) {
      v16h bf = load_frag_bt(wT + (size_t)(n_w + ni * 16) * C_ + k0, C_, lane);
      acc[0][ni] = wmma_f16(a0, bf, acc[0][ni]);
      acc[1][ni] = wmma_f16(a1, bf, acc[1][ni]);
    }
  }

  const int half = (lane >> 4) << 3;
  const int col_l = lane & 15;
#pragma unroll
  for (int mi = 0; mi < 2; ++mi)
#pragma unroll
    for (int ni = 0; ni < 4; ++ni)
#pragma unroll
      for (int r = 0; r < 8; ++r) {
        int m = m_w + mi * 16 + r + half;
        int n = n_w + ni * 16 + col_l;
        out[(size_t)m * C_ + n] = acc[mi][ni][r] + bias[n];
      }
}

// ---------------- host launcher ----------------
extern "C" void kernel_launch(void* const* d_in, const int* in_sizes, int n_in,
                              void* d_out, int out_size, void* d_ws, size_t ws_size,
                              hipStream_t stream) {
  const float* x      = (const float*)d_in[0];
  const float* qkv_w  = (const float*)d_in[1];
  const float* qkv_b  = (const float*)d_in[2];
  const float* out_w  = (const float*)d_in[3];
  const float* out_b  = (const float*)d_in[4];
  float* out = (float*)d_out;

  _Float16* ws = (_Float16*)d_ws;
  size_t off = 0;
  _Float16* xh    = ws + off; off += (size_t)M_ * C_;          // 4096x1024
  _Float16* wqkvT = ws + off; off += (size_t)3 * C_ * C_;      // 3072x1024 (transposed)
  _Float16* woutT = ws + off; off += (size_t)C_ * C_;          // 1024x1024 (transposed)
  _Float16* qh    = ws + off; off += (size_t)B_ * H_ * T_ * DH_;
  _Float16* kh    = ws + off; off += (size_t)B_ * H_ * T_ * DH_;
  _Float16* vTh   = ws + off; off += (size_t)B_ * H_ * T_ * DH_;
  _Float16* attnh = ws + off; off += (size_t)M_ * C_;
  // total: 24M halves = 48 MB of d_ws

  cvt_f32_to_f16_kernel<<<1024, 256, 0, stream>>>(x, xh, (size_t)M_ * C_);
  transpose_cvt_kernel<<<1024, 256, 0, stream>>>(qkv_w, wqkvT, C_, 3 * C_);
  transpose_cvt_kernel<<<512, 256, 0, stream>>>(out_w, woutT, C_, C_);

  qkv_gemm_kernel<<<dim3(3 * C_ / 256, M_ / 64), 256, 0, stream>>>(
      xh, wqkvT, qkv_b, qh, kh, vTh);

  attn_kernel<<<dim3(B_ * H_, T_ / 128), 256, 0, stream>>>(qh, kh, vTh, attnh);

  out_gemm_kernel<<<dim3(C_ / 256, M_ / 64), 256, 0, stream>>>(
      attnh, woutT, out_b, out);
}
